// UHGModel_56255481643281
// MI455X (gfx1250) — matrix operand order repaired and checked by reading, not compiled, not run on hardware
//
#include <hip/hip_runtime.h>
#include <hip/hip_bf16.h>
#include <math.h>

typedef __attribute__((ext_vector_type(16))) __bf16 v16bf;
typedef __attribute__((ext_vector_type(8)))  float  v8f;

#define EPSF 1e-9f

// ---------------------------------------------------------------------------
// Prep: transpose + convert weights to bf16:  Wt[n][k] = bf16(W[k][n]).
// Run once per launch; makes the WMMA B-fragment a contiguous 32B load
// (lane n needs K = half*16 + 0..15 consecutive -> one v16bf vector).
// ---------------------------------------------------------------------------
__global__ __launch_bounds__(256)
void conv_w_bf16t(const float* __restrict__ W, __bf16* __restrict__ Wt,
                  int K, int MOUT)
{
    const int idx = blockIdx.x * 256 + threadIdx.x;     // over K*MOUT
    if (idx >= K * MOUT) return;
    const int k = idx / MOUT;
    const int n = idx % MOUT;
    Wt[(size_t)n * K + k] = (__bf16)W[idx];             // coalesced read
}

// ---------------------------------------------------------------------------
// Fused GEMM (bf16 WMMA, f32 acc) + bias + ReLU + row L2-normalize.
//   H[N,MOUT] = normalize(relu(X[N,K] @ W[K,MOUT] + bias))
// One wave computes one 16x16 tile via v_wmma_f32_16x16x32_bf16; the
// workgroup ((MOUT/16) waves) covers 16 rows x MOUT cols, then normalizes
// rows cooperatively through LDS (norm spans all MOUT columns).
// Weights come pre-transposed/converted as Wt[MOUT][K] bf16.
// ---------------------------------------------------------------------------
template<int K, int MOUT>
__global__ __launch_bounds__((MOUT/16)*32)
void gemm_bias_relu_norm(const float* __restrict__ X,
                         const __bf16* __restrict__ Wt,
                         const float* __restrict__ bias,
                         float* __restrict__ H,
                         int N)
{
    __shared__ float sm[16 * MOUT];

    const int wave = threadIdx.x >> 5;
    const int lane = threadIdx.x & 31;
    const int m    = lane & 15;        // A row within tile / B,C column
    const int half = lane >> 4;        // lane half selects K/M striping
    const int n0   = wave * 16;        // output-column tile base
    const int row0 = blockIdx.x * 16;  // output-row tile base

    int arow = row0 + m;
    if (arow >= N) arow = N - 1;       // clamp (WMMA cannot predicate lanes)
    const float*  __restrict__ xrow = X  + (size_t)arow * K;
    const __bf16* __restrict__ wrow = Wt + (size_t)(n0 + m) * K + half * 16;

    v8f acc = {};
    #pragma unroll
    for (int k0 = 0; k0 < K; k0 += 32) {
        // A-matrix 16x32 bf16 layout (matches B): VGPR v ->
        //   K = half*16 + v*2  (packed pair); per lane this is
        //   one run of 16 consecutive fp32 -> merged b128 loads + pk cvt.
        v16bf a;
        #pragma unroll
        for (int v = 0; v < 8; ++v) {
            const int ka = half * 16 + v * 2;
            const float2 f2 = *(const float2*)(xrow + k0 + ka);
            a[2*v]   = (__bf16)f2.x;
            a[2*v+1] = (__bf16)f2.y;
        }
        // B-matrix 32x16 bf16 layout: lane n, element i -> K = half*16 + i.
        // With Wt[n][k] this is 16 consecutive bf16 = one 32B vector load.
        const v16bf b = *(const v16bf*)(wrow + k0);

        acc = __builtin_amdgcn_wmma_f32_16x16x32_bf16(
                  /*neg_a=*/false, a, /*neg_b=*/false, b,
                  /*c_mod=*/(short)0, acc, /*reuse_a=*/false, /*reuse_b=*/false);
    }

    // C/D layout: VGPR r -> row = half*8 + r, col = lane&15.  bias + ReLU.
    const float bv = bias[n0 + m];
    #pragma unroll
    for (int r = 0; r < 8; ++r) {
        const int mr = half * 8 + r;
        float v = acc[r] + bv;
        sm[mr * MOUT + n0 + m] = v > 0.f ? v : 0.f;
    }
    __syncthreads();

    // Row L2 norm: G = MOUT/8 consecutive threads per row, 8 cols each,
    // reduced with wave32 xor-shuffles (G divides 32).
    constexpr int G = MOUT / 8;
    const int row = threadIdx.x / G;
    const int sub = threadIdx.x % G;
    float vals[8];
    float s = 0.f;
    #pragma unroll
    for (int j = 0; j < 8; ++j) {
        vals[j] = sm[row * MOUT + sub * 8 + j];
        s += vals[j] * vals[j];
    }
    #pragma unroll
    for (int off = G / 2; off > 0; off >>= 1)
        s += __shfl_xor(s, off, 32);
    const float scale = 1.f / (sqrtf(s) + EPSF);

    const int orow = row0 + row;
    if (orow < N) {
        #pragma unroll
        for (int j = 0; j < 8; ++j)
            H[(size_t)orow * MOUT + sub * 8 + j] = vals[j] * scale;
    }
}

// ---------------------------------------------------------------------------
// Message passing: one wave per edge, 4 features/lane (128 = 32*4).
// Minkowski form: + on features 0..126, - on feature 127 (lane 31 .w).
// Gathers are L2-resident (h = 51.2MB < 192MB L2); scatter via f32 atomics.
// ---------------------------------------------------------------------------
__global__ __launch_bounds__(256)
void msg_pass(const float* __restrict__ H,
              const int* __restrict__ rows,
              const int* __restrict__ cols,
              float* __restrict__ AGG,
              int E)
{
    const int e = blockIdx.x * 8 + (threadIdx.x >> 5);
    if (e >= E) return;
    const int lane = threadIdx.x & 31;
    const int r = rows[e];
    const int c = cols[e];

    const float4 xs = *(const float4*)(H + (size_t)r * 128 + lane * 4);
    const float4 xt = *(const float4*)(H + (size_t)c * 128 + lane * 4);
    const float sgn = (lane == 31) ? -1.f : 1.f;   // last coord negated

    float pst = xs.x*xt.x + xs.y*xt.y + xs.z*xt.z + sgn * xs.w*xt.w;
    float pss = xs.x*xs.x + xs.y*xs.y + xs.z*xs.z + sgn * xs.w*xs.w;
    float ptt = xt.x*xt.x + xt.y*xt.y + xt.z*xt.z + sgn * xt.w*xt.w;
    #pragma unroll
    for (int off = 16; off > 0; off >>= 1) {
        pst += __shfl_xor(pst, off, 32);
        pss += __shfl_xor(pss, off, 32);
        ptt += __shfl_xor(ptt, off, 32);
    }

    float q = 1.f - (pst * pst) / (pss * ptt + EPSF);
    q = q > 0.f ? q : 0.f;
    const float w = expf(-sqrtf(q));

    float* dst = AGG + (size_t)c * 128 + lane * 4;
    atomicAdd(dst + 0, xs.x * w);
    atomicAdd(dst + 1, xs.y * w);
    atomicAdd(dst + 2, xs.z * w);
    atomicAdd(dst + 3, xs.w * w);
}

// ---------------------------------------------------------------------------
__global__ void zero_f32(float* __restrict__ p, size_t n)
{
    size_t i = (size_t)blockIdx.x * blockDim.x + threadIdx.x;
    const size_t stride = (size_t)gridDim.x * blockDim.x;
    for (; i < n; i += stride) p[i] = 0.f;
}

// ---------------------------------------------------------------------------
// Classifier head: sigmoid(relu(h @ Wc1 + bc1) @ Wc2 + bc2), h in [N,64].
// Weights cached in LDS; one thread per node (tiny FLOPs, L2-resident).
// ---------------------------------------------------------------------------
__global__ __launch_bounds__(256)
void head(const float* __restrict__ H,
          const float* __restrict__ Wc1, const float* __restrict__ bc1,
          const float* __restrict__ Wc2, const float* __restrict__ bc2,
          float* __restrict__ OUT, int N)
{
    __shared__ float w1[64 * 32];
    __shared__ float b1s[32];
    __shared__ float w2[32];
    for (int i = threadIdx.x; i < 64 * 32; i += 256) w1[i] = Wc1[i];
    if (threadIdx.x < 32) {
        b1s[threadIdx.x] = bc1[threadIdx.x];
        w2[threadIdx.x]  = Wc2[threadIdx.x];
    }
    __syncthreads();

    const int nidx = blockIdx.x * 256 + threadIdx.x;
    if (nidx >= N) return;
    const float* __restrict__ h = H + (size_t)nidx * 64;

    float hv[64];
    #pragma unroll
    for (int i = 0; i < 64; ++i) hv[i] = h[i];

    float z = bc2[0];
    #pragma unroll 4
    for (int j = 0; j < 32; ++j) {
        float t = b1s[j];
        #pragma unroll
        for (int i = 0; i < 64; ++i) t += hv[i] * w1[i * 32 + j];
        t = t > 0.f ? t : 0.f;
        z += t * w2[j];
    }
    OUT[nidx] = 1.f / (1.f + expf(-z));
}

// ---------------------------------------------------------------------------
extern "C" void kernel_launch(void* const* d_in, const int* in_sizes, int n_in,
                              void* d_out, int out_size, void* d_ws, size_t ws_size,
                              hipStream_t stream)
{
    const float* x   = (const float*)d_in[0];
    const int*   ei  = (const int*)  d_in[1];
    const float* Win = (const float*)d_in[2];
    const float* bin = (const float*)d_in[3];
    const float* W1  = (const float*)d_in[4];
    const float* b1  = (const float*)d_in[5];
    const float* W2  = (const float*)d_in[6];
    const float* b2  = (const float*)d_in[7];
    const float* Wc1 = (const float*)d_in[8];
    const float* bc1 = (const float*)d_in[9];
    const float* Wc2 = (const float*)d_in[10];
    const float* bc2 = (const float*)d_in[11];
    float* out = (float*)d_out;

    const int N = in_sizes[0] / 256;   // 100000
    const int E = in_sizes[1] / 2;     // 1600000
    const int* rows = ei;              // edge_index[0]
    const int* cols = ei + E;          // edge_index[1]

    // Workspace layout (all offsets 32B-aligned):
    //   buf0, buf1 : [N,128] f32 ping-pong activations
    //   wt0        : [128][256] bf16 (W_in^T)
    //   wt1        : [128][128] bf16 (W1^T)
    //   wt2        : [ 64][128] bf16 (W2^T)
    float*  buf0 = (float*)d_ws;
    float*  buf1 = buf0 + (size_t)N * 128;
    __bf16* wt0  = (__bf16*)(buf1 + (size_t)N * 128);
    __bf16* wt1  = wt0 + (size_t)128 * 256;
    __bf16* wt2  = wt1 + (size_t)128 * 128;

    const int rtiles = (N + 15) / 16;
    const int eblks  = (E + 7) / 8;

    // Weight prep (tiny): transpose + fp32->bf16.
    conv_w_bf16t<<<(256 * 128 + 255) / 256, 256, 0, stream>>>(Win, wt0, 256, 128);
    conv_w_bf16t<<<(128 * 128 + 255) / 256, 256, 0, stream>>>(W1,  wt1, 128, 128);
    conv_w_bf16t<<<(128 *  64 + 255) / 256, 256, 0, stream>>>(W2,  wt2, 128,  64);

    // h1 = normalize(relu(x @ W_in + b_in))            -> buf0 [N,128]
    gemm_bias_relu_norm<256,128><<<rtiles, 256, 0, stream>>>(x, wt0, bin, buf0, N);
    // agg1 = scatter(h1)                               -> buf1
    zero_f32<<<2048, 256, 0, stream>>>(buf1, (size_t)N * 128);
    msg_pass<<<eblks, 256, 0, stream>>>(buf0, rows, cols, buf1, E);
    // h2 = normalize(relu(agg1 @ W1 + b1))             -> buf0
    gemm_bias_relu_norm<128,128><<<rtiles, 256, 0, stream>>>(buf1, wt1, b1, buf0, N);
    // agg2 = scatter(h2)                               -> buf1
    zero_f32<<<2048, 256, 0, stream>>>(buf1, (size_t)N * 128);
    msg_pass<<<eblks, 256, 0, stream>>>(buf0, rows, cols, buf1, E);
    // h3 = normalize(relu(agg2 @ W2 + b2))             -> buf0 [N,64]
    gemm_bias_relu_norm<128,64><<<rtiles, 128, 0, stream>>>(buf1, wt2, b2, buf0, N);
    // out = sigmoid(relu(h3 @ Wc1 + bc1) @ Wc2 + bc2)
    head<<<(N + 255) / 256, 256, 0, stream>>>(buf0, Wc1, bc1, Wc2, bc2, out, N);
}
